// GraphNVPLayer_80625126081256
// MI455X (gfx1250) — matrix operand order repaired and checked by reading, not compiled, not run on hardware
//
#include <hip/hip_runtime.h>
#include <hip/hip_fp16.h>

// GraphNVP coupling layer for MI455X (gfx1250, wave32).
// f16 data / f32 accum via v_wmma_f32_16x16x32_f16; one wave = one 16-row tile.
// Edge pipeline fused (concat -> e1 -> relu -> e2 -> global_atomic_add_f32
// scatter); edges + f16 nodes + agg fit in the 192MB L2, so HBM traffic is
// just compulsory reads (~80MB) + output writes. Weights are pre-swizzled into
// per-lane WMMA B-fragment order so they load as global_load_b128 (cache-hot,
// no LDS, no block barriers); A fragments come from LDS via ds_load_b128.

typedef __attribute__((ext_vector_type(16))) _Float16 v16h;
typedef __attribute__((ext_vector_type(8)))  _Float16 v8h;
typedef __attribute__((ext_vector_type(8)))  float    v8f;

#define BLOCK 256
#define WPB 8  // waves per block (wave32)

// fragment-ordered f16 weights: [chunk][lane][16 halves]; offsets in v16h units
#define WF_E1 0     // 3 K-chunks (72 -> 96)
#define WF_E2 96    // 1 K-chunk  (16 -> 32)
#define WF_N1 128   // 2 K-chunks (48 -> 64)
#define WF_N2 192   // 1 K-chunk  (16 -> 32)
#define WF_TOTAL 224

#define SX_E 104    // edge sX row stride (halves): 52 dwords -> conflict-free
#define SX_N 72     // node sX row stride: 36 dwords -> conflict-free
#define SH_S 24     // h1 row stride: 12 dwords -> conflict-free

// Intra-wave LDS producer->consumer fence (CDNA5 split counters).
__device__ __forceinline__ void wave_lds_sync() {
  asm volatile("s_wait_dscnt 0" ::: "memory");
}

__device__ __forceinline__ v16h cat8(v8h lo, v8h hi) {
  return __builtin_shufflevector(lo, hi, 0,1,2,3,4,5,6,7,8,9,10,11,12,13,14,15);
}

// ---------------------------------------------------------------- prep kernels

__global__ void zero_f32_kernel(float* __restrict__ p, long long count) {
  long long i = (long long)blockIdx.x * blockDim.x + threadIdx.x;
  if (i < count) p[i] = 0.f;
}

// nodes_m = nodes with column mask_dim zeroed, stored as f16
__global__ void mask_nodes_kernel(const float* __restrict__ nodes,
                                  const int* __restrict__ maskp,
                                  _Float16* __restrict__ dst, long long count) {
  long long i = (long long)blockIdx.x * blockDim.x + threadIdx.x;
  if (i < count) {
    const int col = (int)(i & 31);
    const float v = nodes[i];
    dst[i] = (_Float16)((col == *maskp) ? 0.f : v);
  }
}

// Pack a (fin x 16) f32 weight matrix into WMMA B-fragment order:
// element ((kc*32 + lane)*16 + h) = W[kc*32 + 8*(lane>>4) + h + (h>=8 ? 8 : 0)][lane&15].
__device__ __forceinline__ void pack_frag(const float* __restrict__ W, int fin,
                                          int nchunks, _Float16* __restrict__ dst,
                                          int t) {
  const int total = nchunks * 512;
  for (int idx = t; idx < total; idx += BLOCK) {
    const int h  = idx & 15;
    const int l  = (idx >> 4) & 31;
    const int kc = idx >> 9;
    const int K  = kc * 32 + 8 * (l >> 4) + h + (h >= 8 ? 8 : 0);
    const int n  = l & 15;
    dst[idx] = (_Float16)(K < fin ? W[K * 16 + n] : 0.f);
  }
}

__global__ void pack_weights_kernel(const float* __restrict__ e1W,
                                    const float* __restrict__ e2W,
                                    const float* __restrict__ n1W,
                                    const float* __restrict__ n2W,
                                    _Float16* __restrict__ dst) {
  const int t = threadIdx.x;
  pack_frag(e1W, 72, 3, dst + WF_E1 * 16, t);
  pack_frag(e2W, 16, 1, dst + WF_E2 * 16, t);
  pack_frag(n1W, 48, 2, dst + WF_N1 * 16, t);
  pack_frag(n2W, 16, 1, dst + WF_N2 * 16, t);
}

// ---------------------------------------------------------------- edge kernel
// One wave = 16 edges. h = relu([edge|snd|rcv] @ W1 + b1) @ W2 + b2,
// then atomic scatter-add into agg[receiver]. No block barriers.

__global__ __launch_bounds__(BLOCK) void edge_gn_kernel(
    const float* __restrict__ edges,
    const int* __restrict__ senders,
    const int* __restrict__ receivers,
    const _Float16* __restrict__ nodesm,
    const _Float16* __restrict__ wpack,
    const float* __restrict__ b1,
    const float* __restrict__ b2,
    float* __restrict__ agg,
    int ntiles) {
  __shared__ _Float16 sX[WPB][16][SX_E];  // [edge(8)|snd(32)|rcv(32)|pad0(24)..]
  __shared__ _Float16 sH[WPB][16][SH_S];  // h1 (16x16)
  __shared__ int sSnd[WPB][16];
  __shared__ int sRcv[WPB][16];

  const int tid = threadIdx.x;
  const int wave = tid >> 5;
  const int lane = tid & 31;
  const int tile = blockIdx.x * WPB + wave;
  if (tile >= ntiles) return;              // wave-uniform; no block barriers
  const int e0 = tile * 16;

  // B fragments: two global_load_b128 per fragment, cache-hot in WGP$/L2.
  const v16h* __restrict__ wf = (const v16h*)wpack;
  v16h bw1[3];
#pragma unroll
  for (int kc = 0; kc < 3; ++kc) bw1[kc] = wf[WF_E1 + kc * 32 + lane];
  const v16h bw2 = wf[WF_E2 + lane];

  if (lane < 16) {
    sSnd[wave][lane] = senders[e0 + lane];
    sRcv[wave][lane] = receivers[e0 + lane];
  }
  const v8h zero8 = {};
  for (int i = lane; i < 16 * 3; i += 32)  // zero K-pad cols [72,96)
    *(v8h*)&sX[wave][i / 3][72 + (i % 3) * 8] = zero8;
  wave_lds_sync();

  // fill: 16 rows x 9 16B chunks (1 edge-cvt + 4 sender + 4 receiver)
  for (int i = lane; i < 16 * 9; i += 32) {
    const int r = i / 9, j = i % 9;
    v8h val;
    if (j == 0) {
      const float4* ep = (const float4*)(edges + (size_t)(e0 + r) * 8);
      const float4 f0 = ep[0], f1 = ep[1];
      val[0] = (_Float16)f0.x; val[1] = (_Float16)f0.y;
      val[2] = (_Float16)f0.z; val[3] = (_Float16)f0.w;
      val[4] = (_Float16)f1.x; val[5] = (_Float16)f1.y;
      val[6] = (_Float16)f1.z; val[7] = (_Float16)f1.w;
    } else if (j < 5) {
      val = ((const v8h*)(nodesm + (size_t)sSnd[wave][r] * 32))[j - 1];
    } else {
      val = ((const v8h*)(nodesm + (size_t)sRcv[wave][r] * 32))[j - 5];
    }
    *(v8h*)&sX[wave][r][j * 8] = val;
  }
  wave_lds_sync();

  const int l15 = lane & 15;
  const int grp = lane >> 4;
  const v8h* __restrict__ rowp = (const v8h*)&sX[wave][l15][0];  // 13 v8h/row
  v8f acc = {};
#pragma unroll
  for (int kc = 0; kc < 3; ++kc) {          // A halves: K 8g..8g+7 | 16+8g..23+8g
    const v16h a = cat8(rowp[kc * 4 + grp], rowp[kc * 4 + 2 + grp]);
    acc = __builtin_amdgcn_wmma_f32_16x16x32_f16(false, a, false, bw1[kc],
                                                 (short)0, acc, false, false);
  }
  const float bias1 = b1[l15];
#pragma unroll
  for (int i = 0; i < 8; ++i) {             // C elem: M = i+8*grp, N = l15
    float v = acc[i] + bias1;
    v = v > 0.f ? v : 0.f;                  // relu
    sH[wave][i + 8 * grp][l15] = (_Float16)v;
  }
  wave_lds_sync();

  const v16h a2 = cat8(*(const v8h*)&sH[wave][l15][8 * grp], zero8);
  v8f acc2 = {};
  acc2 = __builtin_amdgcn_wmma_f32_16x16x32_f16(false, a2, false, bw2,
                                                (short)0, acc2, false, false);
  const float bias2 = b2[l15];
#pragma unroll
  for (int i = 0; i < 8; ++i) {
    const int m = i + 8 * grp;
    unsafeAtomicAdd(&agg[(size_t)sRcv[wave][m] * 16 + l15], acc2[i] + bias2);
  }
}

// ---------------------------------------------------------------- node kernel
// One wave = 16 nodes. x = [node(32)|agg(16)] -> n1(act) -> n2(act) -> one
// column of n3 (col 0 for trans head, col mask_dim for scale head).

__global__ __launch_bounds__(BLOCK) void node_gn_kernel(
    const _Float16* __restrict__ nodesm,
    const float* __restrict__ agg,
    const _Float16* __restrict__ wpack,
    const float* __restrict__ b1,
    const float* __restrict__ b2,
    const float* __restrict__ w3,
    const float* __restrict__ b3,
    const int* __restrict__ maskp,
    int use_mask, int out_dim, int act_tanh,
    float* __restrict__ outv,
    int ntiles) {
  __shared__ _Float16 sX[WPB][16][SX_N];  // [node(32)|agg(16)|pad0(16)|..]
  __shared__ _Float16 sH[WPB][16][SH_S];
  __shared__ float    sH2[WPB][16][17];

  const int tid = threadIdx.x;
  const int wave = tid >> 5;
  const int lane = tid & 31;
  const int tile = blockIdx.x * WPB + wave;
  if (tile >= ntiles) return;
  const int n0 = tile * 16;

  const v16h* __restrict__ wf = (const v16h*)wpack;
  v16h bw1[2];
#pragma unroll
  for (int kc = 0; kc < 2; ++kc) bw1[kc] = wf[WF_N1 + kc * 32 + lane];
  const v16h bw2 = wf[WF_N2 + lane];

  const v8h zero8 = {};
  for (int i = lane; i < 16 * 2; i += 32)  // zero K-pad cols [48,64)
    *(v8h*)&sX[wave][i / 2][48 + (i % 2) * 8] = zero8;
  // fill: 16 rows x 6 16B chunks (4 node-f16 + 2 agg-cvt)
  for (int i = lane; i < 16 * 6; i += 32) {
    const int r = i / 6, j = i % 6;
    const size_t node = (size_t)(n0 + r);
    v8h val;
    if (j < 4) {
      val = ((const v8h*)(nodesm + node * 32))[j];
    } else {
      const float4* ap = (const float4*)(agg + node * 16 + (j - 4) * 8);
      const float4 f0 = ap[0], f1 = ap[1];
      val[0] = (_Float16)f0.x; val[1] = (_Float16)f0.y;
      val[2] = (_Float16)f0.z; val[3] = (_Float16)f0.w;
      val[4] = (_Float16)f1.x; val[5] = (_Float16)f1.y;
      val[6] = (_Float16)f1.z; val[7] = (_Float16)f1.w;
    }
    *(v8h*)&sX[wave][r][j * 8] = val;
  }
  wave_lds_sync();

  const int l15 = lane & 15;
  const int grp = lane >> 4;
  const v8h* __restrict__ rowp = (const v8h*)&sX[wave][l15][0];  // 9 v8h/row
  v8f acc = {};
#pragma unroll
  for (int kc = 0; kc < 2; ++kc) {
    const v16h a = cat8(rowp[kc * 4 + grp], rowp[kc * 4 + 2 + grp]);
    acc = __builtin_amdgcn_wmma_f32_16x16x32_f16(false, a, false, bw1[kc],
                                                 (short)0, acc, false, false);
  }
  const float bias1 = b1[l15];
#pragma unroll
  for (int i = 0; i < 8; ++i) {
    float v = acc[i] + bias1;
    v = act_tanh ? tanhf(v) : (v > 0.f ? v : 0.f);
    sH[wave][i + 8 * grp][l15] = (_Float16)v;
  }
  wave_lds_sync();

  const v16h a2 = cat8(*(const v8h*)&sH[wave][l15][8 * grp], zero8);
  v8f acc2 = {};
  acc2 = __builtin_amdgcn_wmma_f32_16x16x32_f16(false, a2, false, bw2,
                                                (short)0, acc2, false, false);
  const float bias2 = b2[l15];
#pragma unroll
  for (int i = 0; i < 8; ++i) {
    float v = acc2[i] + bias2;
    v = act_tanh ? tanhf(v) : (v > 0.f ? v : 0.f);
    sH2[wave][i + 8 * grp][l15] = v;
  }
  wave_lds_sync();

  if (lane < 16) {                          // n3: single column, 16-MAC dot
    const int c = use_mask ? *maskp : 0;
    float dot = b3[c];
#pragma unroll
    for (int k = 0; k < 16; ++k) dot += sH2[wave][lane][k] * w3[k * out_dim + c];
    outv[n0 + lane] = dot;
  }
}

// ---------------------------------------------------------------- finalize
// new_nodes = nodes * exp(scale*mask) + trans*mask ; logdetJ = sum(scale_mask)

__global__ __launch_bounds__(BLOCK) void finalize_kernel(
    const float* __restrict__ nodes,
    const float* __restrict__ transv,
    const float* __restrict__ scalev,
    const int* __restrict__ maskp,
    float* __restrict__ out, int n) {
  __shared__ float red[BLOCK];
  const int i = blockIdx.x * BLOCK + threadIdx.x;
  float s = 0.f;
  if (i < n) {
    const int m = *maskp;
    const float sc = scalev[i];
    s = sc;
    const size_t base = (size_t)i * 32;
#pragma unroll
    for (int j = 0; j < 8; ++j)
      ((float4*)(out + base))[j] = ((const float4*)(nodes + base))[j];
    out[base + m] = nodes[base + m] * expf(sc) + transv[i];
  }
  red[threadIdx.x] = s;
  __syncthreads();
  for (int off = BLOCK / 2; off > 0; off >>= 1) {
    if (threadIdx.x < off) red[threadIdx.x] += red[threadIdx.x + off];
    __syncthreads();
  }
  if (threadIdx.x == 0) unsafeAtomicAdd(&out[(size_t)n * 32], red[0]);
}

// ---------------------------------------------------------------- launch

extern "C" void kernel_launch(void* const* d_in, const int* in_sizes, int n_in,
                              void* d_out, int out_size, void* d_ws, size_t ws_size,
                              hipStream_t stream) {
  const float* nodes     = (const float*)d_in[0];
  const float* edges     = (const float*)d_in[1];
  const int*   senders   = (const int*)d_in[2];
  const int*   receivers = (const int*)d_in[3];
  const int*   maskp     = (const int*)d_in[4];
  const int N = in_sizes[0] / 32;
  const int E = in_sizes[2];

  // params flatten: e1W,e1b,e2W,e2b,n1W,n1b,n2W,n2b,n3W,n3b (trans then scale)
  const float* t_e1W=(const float*)d_in[5];  const float* t_e1b=(const float*)d_in[6];
  const float* t_e2W=(const float*)d_in[7];  const float* t_e2b=(const float*)d_in[8];
  const float* t_n1W=(const float*)d_in[9];  const float* t_n1b=(const float*)d_in[10];
  const float* t_n2W=(const float*)d_in[11]; const float* t_n2b=(const float*)d_in[12];
  const float* t_n3W=(const float*)d_in[13]; const float* t_n3b=(const float*)d_in[14];
  const float* s_e1W=(const float*)d_in[15]; const float* s_e1b=(const float*)d_in[16];
  const float* s_e2W=(const float*)d_in[17]; const float* s_e2b=(const float*)d_in[18];
  const float* s_n1W=(const float*)d_in[19]; const float* s_n1b=(const float*)d_in[20];
  const float* s_n2W=(const float*)d_in[21]; const float* s_n2b=(const float*)d_in[22];
  const float* s_n3W=(const float*)d_in[23]; const float* s_n3b=(const float*)d_in[24];

  // workspace carve-up (~21 MB)
  char* ws = (char*)d_ws;
  size_t off = 0;
  auto carve = [&](size_t bytes) -> char* {
    char* p = ws + off;
    off += (bytes + 255) & ~(size_t)255;
    return p;
  };
  float*     aggT    = (float*)carve((size_t)N * 16 * 4);
  float*     aggS    = (float*)carve((size_t)N * 16 * 4);
  float*     transO  = (float*)carve((size_t)N * 4);
  float*     scaleO  = (float*)carve((size_t)N * 4);
  _Float16*  nodesmH = (_Float16*)carve((size_t)N * 32 * 2);
  _Float16*  wpT     = (_Float16*)carve((size_t)WF_TOTAL * 16 * 2);
  _Float16*  wpS     = (_Float16*)carve((size_t)WF_TOTAL * 16 * 2);
  (void)ws_size; (void)n_in; (void)out_size;

  // 1) zero accumulators + logdetJ slot (harness does not re-zero)
  {
    long long c = (long long)N * 16;
    int blocks = (int)((c + BLOCK - 1) / BLOCK);
    zero_f32_kernel<<<blocks, BLOCK, 0, stream>>>(aggT, c);
    zero_f32_kernel<<<blocks, BLOCK, 0, stream>>>(aggS, c);
    zero_f32_kernel<<<1, BLOCK, 0, stream>>>((float*)d_out + (size_t)N * 32, 1);
  }
  // 2) masked nodes -> f16
  {
    long long c = (long long)N * 32;
    mask_nodes_kernel<<<(int)((c + BLOCK - 1) / BLOCK), BLOCK, 0, stream>>>(
        nodes, maskp, nodesmH, c);
  }
  // 3) pack weights into B-fragment order
  pack_weights_kernel<<<1, BLOCK, 0, stream>>>(t_e1W, t_e2W, t_n1W, t_n2W, wpT);
  pack_weights_kernel<<<1, BLOCK, 0, stream>>>(s_e1W, s_e2W, s_n1W, s_n2W, wpS);
  // 4) fused edge MLP + scatter-add (both heads)
  {
    const int ntiles = (E + 15) / 16;
    const int blocks = (ntiles + WPB - 1) / WPB;
    edge_gn_kernel<<<blocks, BLOCK, 0, stream>>>(edges, senders, receivers,
                                                 nodesmH, wpT, t_e1b, t_e2b, aggT, ntiles);
    edge_gn_kernel<<<blocks, BLOCK, 0, stream>>>(edges, senders, receivers,
                                                 nodesmH, wpS, s_e1b, s_e2b, aggS, ntiles);
  }
  // 5) node MLPs (trans: relu head / col 0 ; scale: tanh head / col mask_dim)
  {
    const int ntiles = (N + 15) / 16;
    const int blocks = (ntiles + WPB - 1) / WPB;
    node_gn_kernel<<<blocks, BLOCK, 0, stream>>>(nodesmH, aggT, wpT, t_n1b, t_n2b,
                                                 t_n3W, t_n3b, maskp, 0, 1, 0, transO, ntiles);
    node_gn_kernel<<<blocks, BLOCK, 0, stream>>>(nodesmH, aggS, wpS, s_n1b, s_n2b,
                                                 s_n3W, s_n3b, maskp, 1, 32, 1, scaleO, ntiles);
  }
  // 6) assemble outputs + logdetJ
  finalize_kernel<<<(N + BLOCK - 1) / BLOCK, BLOCK, 0, stream>>>(
      nodes, transO, scaleO, maskp, (float*)d_out, N);
}
